// GraphSAGE_36447092474036
// MI455X (gfx1250) — compile-verified, matrix operand order checked by reading, not run on hardware
//
#include <hip/hip_runtime.h>
#include <hip/hip_bf16.h>
#include <math.h>

#define N_NODES 100000
#define N_EDGES 1600000
#define DIM_IN  64
#define DIM_H   128
#define DIM_OUT 8

typedef __attribute__((ext_vector_type(16))) __bf16 v16bf;
typedef __attribute__((ext_vector_type(8)))  __bf16 v8bf;
typedef __attribute__((ext_vector_type(8)))  float  v8f;

// ---------------------------------------------------------------------------
// Degree: cnt[dst] += 1  (f32 atomics; cnt buffer pre-zeroed)
// ---------------------------------------------------------------------------
__global__ __launch_bounds__(256) void k_degree(const int* __restrict__ dst,
                                                float* __restrict__ cnt, int ne) {
    int i = blockIdx.x * 256 + threadIdx.x;
    if (i < ne)
        __hip_atomic_fetch_add(&cnt[dst[i]], 1.0f, __ATOMIC_RELAXED,
                               __HIP_MEMORY_SCOPE_AGENT);
}

// cnt -> 1/max(cnt,1), in place
__global__ __launch_bounds__(256) void k_invert(float* __restrict__ c, int n) {
    int i = blockIdx.x * 256 + threadIdx.x;
    if (i < n) c[i] = 1.0f / fmaxf(c[i], 1.0f);
}

// ---------------------------------------------------------------------------
// Scatter-add of feature rows: agg[dst] += feat[src], float4 chunks per thread
// ---------------------------------------------------------------------------
__global__ __launch_bounds__(256) void k_scatter(const float* __restrict__ feat,
                                                 const int* __restrict__ src,
                                                 const int* __restrict__ dst,
                                                 float* __restrict__ agg,
                                                 int ne, int dim) {
    int cpe = dim >> 2;                       // float4 chunks per edge
    int idx = blockIdx.x * 256 + threadIdx.x;
    if (idx >= ne * cpe) return;
    int e = idx / cpe;
    int c = idx - e * cpe;
    int s = src[e], d = dst[e];
    const float4 v = *(const float4*)(feat + (size_t)s * dim + c * 4);
    float* o = agg + (size_t)d * dim + c * 4;
    __hip_atomic_fetch_add(o + 0, v.x, __ATOMIC_RELAXED, __HIP_MEMORY_SCOPE_AGENT);
    __hip_atomic_fetch_add(o + 1, v.y, __ATOMIC_RELAXED, __HIP_MEMORY_SCOPE_AGENT);
    __hip_atomic_fetch_add(o + 2, v.z, __ATOMIC_RELAXED, __HIP_MEMORY_SCOPE_AGENT);
    __hip_atomic_fetch_add(o + 3, v.w, __ATOMIC_RELAXED, __HIP_MEMORY_SCOPE_AGENT);
}

// scalar scatter for layer-4 (dim==1)
__global__ __launch_bounds__(256) void k_scatter1(const float* __restrict__ h,
                                                  const int* __restrict__ src,
                                                  const int* __restrict__ dst,
                                                  float* __restrict__ agg, int ne) {
    int i = blockIdx.x * 256 + threadIdx.x;
    if (i < ne)
        __hip_atomic_fetch_add(&agg[dst[i]], h[src[i]], __ATOMIC_RELAXED,
                               __HIP_MEMORY_SCOPE_AGENT);
}

// ---------------------------------------------------------------------------
// Build bf16 A = [mean | x]  (K = 2*di per node, row-major)
// ---------------------------------------------------------------------------
__global__ __launch_bounds__(256) void k_buildA(const float* __restrict__ agg,
                                                const float* __restrict__ feat,
                                                const float* __restrict__ inv,
                                                __bf16* __restrict__ A,
                                                int di, int n_nodes) {
    int ktot = 2 * di;
    int idx = blockIdx.x * 256 + threadIdx.x;
    if (idx >= n_nodes * ktot) return;
    int n = idx / ktot;
    int k = idx - n * ktot;
    float v = (k < di) ? agg[(size_t)n * di + k] * inv[n]
                       : feat[(size_t)n * di + (k - di)];
    A[idx] = (__bf16)v;
}

// ---------------------------------------------------------------------------
// Pack B = [Wl; Wr] (2*di x do, f32 row-major) into per-lane WMMA fragment
// order:  Wp[ ((tile*ksteps + s)*32 + lane)*16 + e ]
// lane: hlf=lane>>4, n = tile*16 + (lane&15)
// e<8 : k = s*32 + hlf*8 + e ;  e>=8 : k = s*32 + 16 + hlf*8 + (e-8)
// (mirrors the 16-bit A-matrix lane layout, ISA 7.12.2)
// ---------------------------------------------------------------------------
__global__ __launch_bounds__(256) void k_packW(const float* __restrict__ Wl,
                                               const float* __restrict__ Wr,
                                               __bf16* __restrict__ Wp,
                                               int di, int dout, int ksteps) {
    int total = (dout >> 4) * ksteps * 32 * 16;
    int idx = blockIdx.x * 256 + threadIdx.x;
    if (idx >= total) return;
    int e    = idx & 15;
    int lane = (idx >> 4) & 31;
    int s    = (idx >> 9) % ksteps;
    int t    = idx / (ksteps << 9);
    int hlf  = lane >> 4, l16 = lane & 15;
    int klocal = (e < 8) ? (hlf * 8 + e) : (16 + hlf * 8 + (e - 8));
    int k = s * 32 + klocal;
    int n = t * 16 + l16;
    float v = (k < di) ? Wl[k * dout + n] : Wr[(k - di) * dout + n];
    Wp[idx] = (__bf16)v;
}

// ---------------------------------------------------------------------------
// WMMA GEMM: out[M x 128] = A[M x Ktot](bf16) * Wp + bias, optional relu.
// Block = 256 threads = 8 waves; wave w owns output col-tile [16w,16w+16).
// blockIdx.x owns row-tile [16*bx, 16*bx+16).  M % 16 == 0 (100000).
// ---------------------------------------------------------------------------
__global__ __launch_bounds__(256) void k_gemm_wmma(const __bf16* __restrict__ A,
                                                   const __bf16* __restrict__ Wp,
                                                   const float* __restrict__ bias,
                                                   float* __restrict__ out,
                                                   int ktot, int ncols, int relu) {
    const int lane = threadIdx.x & 31;
    const int wave = threadIdx.x >> 5;          // col tile id (0..7)
    const int m0   = blockIdx.x * 16;
    const int n0   = wave * 16;
    const int hlf  = lane >> 4;
    const int l16  = lane & 15;
    const int ksteps = ktot >> 5;

    const int row = m0 + l16;
    const float bv = bias[n0 + l16];
    v8f c;
#pragma unroll
    for (int j = 0; j < 8; ++j) c[j] = bv;

    const __bf16* arow = A + (size_t)row * ktot + hlf * 8;
    const __bf16* wp   = Wp + ((size_t)wave * ksteps * 32 + lane) * 16;

    for (int s = 0; s < ksteps; ++s) {
        v8bf alo = *(const v8bf*)(arow + s * 32);
        v8bf ahi = *(const v8bf*)(arow + s * 32 + 16);
        v16bf a;
#pragma unroll
        for (int i = 0; i < 8; ++i) { a[i] = alo[i]; a[i + 8] = ahi[i]; }
        v16bf b = *(const v16bf*)(wp + (size_t)s * 32 * 16);
        c = __builtin_amdgcn_wmma_f32_16x16x32_bf16(
                false, a, false, b, (short)0, c, false, false);
    }
    if (relu) {
#pragma unroll
        for (int j = 0; j < 8; ++j) c[j] = fmaxf(c[j], 0.0f);
    }
    float* orow = out + (size_t)(m0 + hlf * 8) * ncols + n0 + l16;
#pragma unroll
    for (int j = 0; j < 8; ++j) orow[(size_t)j * ncols] = c[j];
}

// ---------------------------------------------------------------------------
// Layer 3 (128 -> 1): wave-per-node dot products; emits h3 + scalar_field out
// ---------------------------------------------------------------------------
__global__ __launch_bounds__(256) void k_layer3(const float* __restrict__ agg,
                                                const float* __restrict__ h2,
                                                const float* __restrict__ inv,
                                                const float* __restrict__ W3l,
                                                const float* __restrict__ W3r,
                                                const float* __restrict__ b3,
                                                float* __restrict__ h3,
                                                float* __restrict__ sf_out,
                                                int n_nodes) {
    int wave = threadIdx.x >> 5, lane = threadIdx.x & 31;
    int n = blockIdx.x * 8 + wave;
    if (n >= n_nodes) return;
    float iv = inv[n];
    float acc = 0.0f;
    const float* ar = agg + (size_t)n * DIM_H;
    const float* hr = h2 + (size_t)n * DIM_H;
#pragma unroll
    for (int k = lane; k < DIM_H; k += 32)
        acc += ar[k] * iv * W3l[k] + hr[k] * W3r[k];
#pragma unroll
    for (int off = 16; off > 0; off >>= 1) acc += __shfl_xor(acc, off, 32);
    if (lane == 0) {
        float v = acc + b3[0];
        h3[n] = v;
        sf_out[n] = v;
    }
}

// ---------------------------------------------------------------------------
// Layer 4 (1 -> 8) + log_softmax, thread-per-node
// ---------------------------------------------------------------------------
__global__ __launch_bounds__(256) void k_layer4(const float* __restrict__ agg4,
                                                const float* __restrict__ h3,
                                                const float* __restrict__ inv,
                                                const float* __restrict__ W4l,
                                                const float* __restrict__ W4r,
                                                const float* __restrict__ b4,
                                                float* __restrict__ out,
                                                int n_nodes) {
    int n = blockIdx.x * 256 + threadIdx.x;
    if (n >= n_nodes) return;
    float m = agg4[n] * inv[n];
    float s = h3[n];
    float v[DIM_OUT];
    float mx = -INFINITY;
#pragma unroll
    for (int o = 0; o < DIM_OUT; ++o) {
        v[o] = m * W4l[o] + s * W4r[o] + b4[o];
        mx = fmaxf(mx, v[o]);
    }
    float sum = 0.0f;
#pragma unroll
    for (int o = 0; o < DIM_OUT; ++o) sum += __expf(v[o] - mx);
    float lse = mx + __logf(sum);
    float* orow = out + (size_t)n * DIM_OUT;
#pragma unroll
    for (int o = 0; o < DIM_OUT; ++o) orow[o] = v[o] - lse;
}

// ---------------------------------------------------------------------------
extern "C" void kernel_launch(void* const* d_in, const int* in_sizes, int n_in,
                              void* d_out, int out_size, void* d_ws, size_t ws_size,
                              hipStream_t stream) {
    const float* x   = (const float*)d_in[0];
    const int*   ei  = (const int*)d_in[1];
    const int*   src = ei;
    const int*   dst = ei + N_EDGES;
    const float* W1l = (const float*)d_in[2];
    const float* W1r = (const float*)d_in[3];
    const float* b1  = (const float*)d_in[4];
    const float* W2l = (const float*)d_in[5];
    const float* W2r = (const float*)d_in[6];
    const float* b2  = (const float*)d_in[7];
    const float* W3l = (const float*)d_in[8];
    const float* W3r = (const float*)d_in[9];
    const float* b3  = (const float*)d_in[10];
    const float* W4l = (const float*)d_in[11];
    const float* W4r = (const float*)d_in[12];
    const float* b4  = (const float*)d_in[13];
    float* out = (float*)d_out;                 // [0,800000) rock_unit, [800000,900000) scalar_field

    // ---- workspace carve (all offsets 16B-aligned) ----
    char* p = (char*)d_ws;
    float* inv  = (float*)p; p += (size_t)N_NODES * 4;              // degree -> 1/deg
    float* agg  = (float*)p; p += (size_t)N_NODES * DIM_H * 4;      // scatter accumulator
    float* h1   = (float*)p; p += (size_t)N_NODES * DIM_H * 4;
    float* h2   = (float*)p; p += (size_t)N_NODES * DIM_H * 4;
    float* h3   = (float*)p; p += (size_t)N_NODES * 4;
    float* agg4 = (float*)p; p += (size_t)N_NODES * 4;
    __bf16* Ab  = (__bf16*)p; p += (size_t)N_NODES * 256 * 2;       // bf16 [mean|x]
    __bf16* Wp  = (__bf16*)p; p += 32768 * 2;                       // packed weights

    const int NB = (N_NODES + 255) / 256;
    const int EB = (N_EDGES + 255) / 256;
    const int MT = N_NODES / 16;                                    // 6250 row tiles

    // ---- degree (shared by all layers) ----
    hipMemsetAsync(inv, 0, (size_t)N_NODES * 4, stream);
    k_degree<<<EB, 256, 0, stream>>>(dst, inv, N_EDGES);
    k_invert<<<NB, 256, 0, stream>>>(inv, N_NODES);

    // ---- layer 1: 64 -> 128, relu ----
    hipMemsetAsync(agg, 0, (size_t)N_NODES * DIM_IN * 4, stream);
    k_scatter<<<(N_EDGES * (DIM_IN / 4) + 255) / 256, 256, 0, stream>>>(
        x, src, dst, agg, N_EDGES, DIM_IN);
    k_buildA<<<(N_NODES * 2 * DIM_IN + 255) / 256, 256, 0, stream>>>(
        agg, x, inv, Ab, DIM_IN, N_NODES);
    k_packW<<<(8 * 4 * 512 + 255) / 256, 256, 0, stream>>>(W1l, W1r, Wp, DIM_IN, DIM_H, 4);
    k_gemm_wmma<<<MT, 256, 0, stream>>>(Ab, Wp, b1, h1, 2 * DIM_IN, DIM_H, 1);

    // ---- layer 2: 128 -> 128, relu ----
    hipMemsetAsync(agg, 0, (size_t)N_NODES * DIM_H * 4, stream);
    k_scatter<<<(N_EDGES * (DIM_H / 4) + 255) / 256, 256, 0, stream>>>(
        h1, src, dst, agg, N_EDGES, DIM_H);
    k_buildA<<<(N_NODES * 2 * DIM_H + 255) / 256, 256, 0, stream>>>(
        agg, h1, inv, Ab, DIM_H, N_NODES);
    k_packW<<<(8 * 8 * 512 + 255) / 256, 256, 0, stream>>>(W2l, W2r, Wp, DIM_H, DIM_H, 8);
    k_gemm_wmma<<<MT, 256, 0, stream>>>(Ab, Wp, b2, h2, 2 * DIM_H, DIM_H, 1);

    // ---- layer 3: 128 -> 1 (wave-reduction), scalar_field output ----
    hipMemsetAsync(agg, 0, (size_t)N_NODES * DIM_H * 4, stream);
    k_scatter<<<(N_EDGES * (DIM_H / 4) + 255) / 256, 256, 0, stream>>>(
        h2, src, dst, agg, N_EDGES, DIM_H);
    k_layer3<<<(N_NODES + 7) / 8, 256, 0, stream>>>(
        agg, h2, inv, W3l, W3r, b3, h3, out + (size_t)N_NODES * DIM_OUT, N_NODES);

    // ---- layer 4: 1 -> 8 + log_softmax ----
    hipMemsetAsync(agg4, 0, (size_t)N_NODES * 4, stream);
    k_scatter1<<<EB, 256, 0, stream>>>(h3, src, dst, agg4, N_EDGES);
    k_layer4<<<NB, 256, 0, stream>>>(agg4, h3, inv, W4l, W4r, b4, out, N_NODES);

    (void)in_sizes; (void)n_in; (void)out_size; (void)ws_size;
}
